// PointConv_15006615734068
// MI455X (gfx1250) — matrix operand order repaired and checked by reading, not compiled, not run on hardware
//
#include <hip/hip_runtime.h>

// ---------------------------------------------------------------------------
// PointNet++-style segmentation net for MI455X (gfx1250).
// All matmuls go through v_wmma_f32_16x16x32_f16 (f16 in / f32 acc).
// Register-blocked: GEMM waves own 64x32 output blocks (8 WMMA / 6 frag loads
// per K-step, ~21 FLOP/byte from L2); point-conv waves fuse both M-tiles per
// B-fragment load.
// ---------------------------------------------------------------------------

typedef __attribute__((ext_vector_type(16))) _Float16 v16h;
typedef __attribute__((ext_vector_type(8)))  _Float16 v8h;
typedef __attribute__((ext_vector_type(8)))  float    v8f;

union V16 { v16h v; v8h h[2]; };

#define BATCH 8
#define NPTS  1024
#define ROWS  (BATCH*NPTS)      // 8192
#define SNB   32
#define NCLS  40

// ---------------------------------------------------------------------------
// Weight prep: f32 (Cout x Kin) -> f16 (CoutPad x Kpad, zero padded), and fold
// bias+BN into per-channel affine:  y = relu(acc*Ac + Bc).
// ---------------------------------------------------------------------------
__global__ void k_prep_weights(const float* __restrict__ W, const float* __restrict__ b,
                               const float* __restrict__ beta, const float* __restrict__ g,
                               const float* __restrict__ m, const float* __restrict__ v,
                               _Float16* __restrict__ Wh, float* __restrict__ Ac,
                               float* __restrict__ Bc,
                               int cout, int coutPad, int kin, int kpad, int has_bn)
{
  int i = blockIdx.x * 256 + threadIdx.x;
  int total = coutPad * kpad;
  if (i < total) {
    int o = i / kpad, k = i % kpad;
    float w = (o < cout && k < kin) ? W[o * kin + k] : 0.f;
    Wh[i] = (_Float16)w;
  }
  if (i < coutPad) {
    if (i < cout) {
      if (has_bn) {
        float s = g[i] * rsqrtf(v[i] + 1e-5f);
        Ac[i] = s;
        Bc[i] = beta[i] + (b[i] - m[i]) * s;
      } else {
        Ac[i] = 1.f; Bc[i] = b[i];
      }
    } else { Ac[i] = 0.f; Bc[i] = 0.f; }
  }
}

// x (B,3,N) -> xyz (B,N,3)
__global__ void k_xyz(const float* __restrict__ x, float* __restrict__ xyz)
{
  int i = blockIdx.x * 256 + threadIdx.x;
  if (i < ROWS) {
    int b = i / NPTS, n = i % NPTS;
    const float* xb = x + (size_t)b * 3 * NPTS;
    float* o = xyz + (size_t)i * 3;
    o[0] = xb[n]; o[1] = xb[NPTS + n]; o[2] = xb[2 * NPTS + n];
  }
}

// ---------------------------------------------------------------------------
// Point conv: one workgroup (8 waves) per point. Build h (32 x KPAD) in LDS
// as f16: [nb-ctr (F) | ctr (F) | density | 0 pad]. Each wave handles N-tiles
// with BOTH M-tiles per B-fragment load (2 WMMA per weight fetch), BN+ReLU
// epilogue, per-channel max over the 32 neighbors via LDS int atomicMax.
// ---------------------------------------------------------------------------
template<int F, int KPAD, int COUT>
__global__ __launch_bounds__(256)
void k_point_conv(const float* __restrict__ feats, const int* __restrict__ idx,
                  const _Float16* __restrict__ Wh, const float* __restrict__ Ac,
                  const float* __restrict__ Bc, float* __restrict__ out)
{
  constexpr int LDSP = KPAD + 8;              // +8 halves: bank-conflict pad
  __shared__ __align__(16) _Float16 hmat[SNB * LDSP];
  __shared__ float psum[SNB * 8];
  __shared__ int   omax[COUT];

  const int pt = blockIdx.x;                  // b*N + n
  const int b  = pt / NPTS;
  const int t  = threadIdx.x;
  const int lane = t & 31, wave = t >> 5;

  for (int o = t; o < COUT; o += 256) omax[o] = 0;   // relu output >= 0

  // ---- build h -----------------------------------------------------------
  {
    const int s = t >> 3, sub = t & 7;        // 8 threads per neighbor row
    const int nb = b * NPTS + idx[(size_t)pt * SNB + s];
    const float* ctr = feats + (size_t)pt * F;
    const float* nbf = feats + (size_t)nb * F;
    float acc = 0.f;
    for (int c = sub; c < F; c += 8) {
      float cv = ctr[c];
      float d  = nbf[c] - cv;
      acc += d * d;
      hmat[s * LDSP + c]     = (_Float16)d;
      hmat[s * LDSP + F + c] = (_Float16)cv;
    }
    psum[s * 8 + sub] = acc;
  }
  __syncthreads();
  {
    const int s = t >> 3, sub = t & 7;
    if (sub == 0) {
      float tot = 0.f;
      #pragma unroll
      for (int j = 0; j < 8; ++j) tot += psum[s * 8 + j];
      float dens = 1.0f / (sqrtf(tot + 1e-12f) + 1e-8f);
      hmat[s * LDSP + 2 * F] = (_Float16)dens;
    }
    for (int k = 2 * F + 1 + sub; k < KPAD; k += 8)
      hmat[s * LDSP + k] = (_Float16)0.f;
  }
  __syncthreads();

  // ---- WMMA: per N-tile, both M-tiles share the B fragment ---------------
  constexpr int NTT = COUT / 16;
  const int half = lane >> 4, lidx = lane & 15;
  for (int nt = wave; nt < NTT; nt += 8) {    // wave-uniform
    const int orow = nt * 16 + lidx;          // output channel
    const _Float16* wrow  = Wh + (size_t)orow * KPAD;
    const _Float16* hrow0 = hmat + lidx * LDSP;          // s rows 0..15
    const _Float16* hrow1 = hmat + (16 + lidx) * LDSP;   // s rows 16..31
    v8f acc0 = {}, acc1 = {};
    for (int k0 = 0; k0 < KPAD; k0 += 32) {
      V16 a0, a1, bb;
      const int ka = k0 + half * 8;           // A: {0-7,16-23} / {8-15,24-31}
      a0.h[0] = *(const v8h*)(hrow0 + ka);
      a0.h[1] = *(const v8h*)(hrow0 + ka + 16);
      a1.h[0] = *(const v8h*)(hrow1 + ka);
      a1.h[1] = *(const v8h*)(hrow1 + ka + 16);
      const int kb = k0 + half * 8;           // B: {0-7,16-23} / {8-15,24-31}
      bb.h[0] = *(const v8h*)(wrow + kb);
      bb.h[1] = *(const v8h*)(wrow + kb + 16);
      acc0 = __builtin_amdgcn_wmma_f32_16x16x32_f16(false, a0.v, false, bb.v,
                                                    (short)0, acc0, false, false);
      acc1 = __builtin_amdgcn_wmma_f32_16x16x32_f16(false, a1.v, false, bb.v,
                                                    (short)0, acc1, false, false);
    }
    const float sc = Ac[orow], sh = Bc[orow];
    float vmax = 0.f;
    #pragma unroll
    for (int r = 0; r < 8; ++r) {
      float v0 = acc0[r] * sc + sh;
      float v1 = acc1[r] * sc + sh;
      v0 = v0 > 0.f ? v0 : 0.f;
      v1 = v1 > 0.f ? v1 : 0.f;
      float vv = v0 > v1 ? v0 : v1;
      vmax = vmax > vv ? vmax : vv;
    }
    atomicMax(&omax[orow], __float_as_int(vmax));  // valid: values >= 0
  }
  __syncthreads();
  for (int o = t; o < COUT; o += 256)
    out[(size_t)pt * COUT + o] = __int_as_float(omax[o]);
}

// ---------------------------------------------------------------------------
// Brute-force 3-NN over the 1024 xyz points of each batch (LDS tiled),
// producing normalized inverse-distance weights. Shared by all feat_props.
// ---------------------------------------------------------------------------
__global__ __launch_bounds__(256)
void k_knn3(const float* __restrict__ xyz, int* __restrict__ knn_idx,
            float* __restrict__ knn_w)
{
  __shared__ float sx[256 * 3];
  const int b = blockIdx.y;
  const int n = blockIdx.x * 256 + threadIdx.x;
  const int row = b * NPTS + n;
  const float cx = xyz[row * 3], cy = xyz[row * 3 + 1], cz = xyz[row * 3 + 2];
  float d0 = 3.4e38f, d1 = 3.4e38f, d2 = 3.4e38f;
  int   i0 = 0, i1 = 0, i2 = 0;
  for (int base = 0; base < NPTS; base += 256) {
    const float* p = xyz + (size_t)(b * NPTS + base + threadIdx.x) * 3;
    sx[threadIdx.x * 3]     = p[0];
    sx[threadIdx.x * 3 + 1] = p[1];
    sx[threadIdx.x * 3 + 2] = p[2];
    __syncthreads();
    for (int jj = 0; jj < 256; ++jj) {
      float dx = sx[jj * 3] - cx, dy = sx[jj * 3 + 1] - cy, dz = sx[jj * 3 + 2] - cz;
      float d = dx * dx + dy * dy + dz * dz;
      int gj = base + jj;
      if (d < d0)      { d2 = d1; i2 = i1; d1 = d0; i1 = i0; d0 = d; i0 = gj; }
      else if (d < d1) { d2 = d1; i2 = i1; d1 = d;  i1 = gj; }
      else if (d < d2) { d2 = d;  i2 = gj; }
    }
    __syncthreads();
  }
  float w0 = 1.f / (d0 + 1e-8f), w1 = 1.f / (d1 + 1e-8f), w2 = 1.f / (d2 + 1e-8f);
  float ws = w0 + w1 + w2;
  knn_w[row * 3]     = w0 / ws;
  knn_w[row * 3 + 1] = w1 / ws;
  knn_w[row * 3 + 2] = w2 / ws;
  knn_idx[row * 3]     = b * NPTS + i0;
  knn_idx[row * 3 + 1] = b * NPTS + i1;
  knn_idx[row * 3 + 2] = b * NPTS + i2;
}

// h f16 = [ pts1 (c1) | 3-NN weighted interp of pts2 (c2) | 0 pad ]
__global__ void k_interp_concat(const float* __restrict__ pts1, int c1,
                                const float* __restrict__ pts2, int c2,
                                const int* __restrict__ knn_idx,
                                const float* __restrict__ knn_w,
                                _Float16* __restrict__ out16, int kpad)
{
  size_t i = (size_t)blockIdx.x * 256 + threadIdx.x;
  size_t total = (size_t)ROWS * kpad;
  if (i >= total) return;
  int row = (int)(i / kpad), k = (int)(i % kpad);
  float v;
  if (k < c1) {
    v = pts1[(size_t)row * c1 + k];
  } else if (k < c1 + c2) {
    int c = k - c1;
    const int*   id = knn_idx + row * 3;
    const float* w  = knn_w   + row * 3;
    v = w[0] * pts2[(size_t)id[0] * c2 + c]
      + w[1] * pts2[(size_t)id[1] * c2 + c]
      + w[2] * pts2[(size_t)id[2] * c2 + c];
  } else v = 0.f;
  out16[i] = (_Float16)v;
}

// ---------------------------------------------------------------------------
// Row GEMM: (8192 x kpad) f16 @ (coutPad x kpad)^T f16 -> f32 acc,
// per-channel affine (+optional relu); optional f32 and/or f16 outputs.
// Each wave computes a 64x32 output block: 4 M-tiles x 2 N-tiles, so each
// K-step does 8 WMMAs on 4 A-frag + 2 B-frag loads (frag reuse 8/6).
// ---------------------------------------------------------------------------
#define GM 4
#define GN 2
__global__ __launch_bounds__(256)
void k_gemm(const _Float16* __restrict__ X, const _Float16* __restrict__ Wh,
            const float* __restrict__ Ac, const float* __restrict__ Bc,
            float* __restrict__ outf, _Float16* __restrict__ out16,
            int kpad, int coutPad, int cout, int npad16, int relu)
{
  const int t = threadIdx.x, lane = t & 31, wave = t >> 5;
  const int nbt = coutPad >> 5;                    // 32-wide N blocks
  const int wt = blockIdx.x * 8 + wave;            // wave-tile id (uniform)
  if (wt >= (ROWS / 64) * nbt) return;
  const int mB = wt / nbt, nB = wt % nbt;
  const int half = lane >> 4, lidx = lane & 15;

  v8f acc[GM][GN] = {};
  const _Float16* xbase = X  + (size_t)(mB * 64 + lidx) * kpad;
  const _Float16* wbase = Wh + (size_t)(nB * 32 + lidx) * kpad;

  for (int k0 = 0; k0 < kpad; k0 += 32) {
    V16 a[GM], bb[GN];
    const int ka = k0 + half * 8;                  // A: {0-7,16-23}/{8-15,24-31}
    #pragma unroll
    for (int mi = 0; mi < GM; ++mi) {
      const _Float16* p = xbase + (size_t)mi * 16 * kpad + ka;
      a[mi].h[0] = *(const v8h*)(p);
      a[mi].h[1] = *(const v8h*)(p + 16);
    }
    const int kb = k0 + half * 8;                  // B: {0-7,16-23}/{8-15,24-31}
    #pragma unroll
    for (int ni = 0; ni < GN; ++ni) {
      const _Float16* p = wbase + (size_t)ni * 16 * kpad + kb;
      bb[ni].h[0] = *(const v8h*)(p);
      bb[ni].h[1] = *(const v8h*)(p + 16);
    }
    #pragma unroll
    for (int mi = 0; mi < GM; ++mi)
      #pragma unroll
      for (int ni = 0; ni < GN; ++ni)
        acc[mi][ni] = __builtin_amdgcn_wmma_f32_16x16x32_f16(
            false, a[mi].v, false, bb[ni].v, (short)0, acc[mi][ni], false, false);
  }

  #pragma unroll
  for (int ni = 0; ni < GN; ++ni) {
    const int orow = nB * 32 + ni * 16 + lidx;
    const float sc = Ac[orow], sh = Bc[orow];
    #pragma unroll
    for (int mi = 0; mi < GM; ++mi) {
      #pragma unroll
      for (int r = 0; r < 8; ++r) {
        const int row = mB * 64 + mi * 16 + r + half * 8;  // C-frag row map
        float v = acc[mi][ni][r] * sc + sh;
        if (relu) v = v > 0.f ? v : 0.f;
        if (outf  && orow < cout) outf[(size_t)row * cout + orow] = v;
        if (out16)                out16[(size_t)row * npad16 + orow] = (_Float16)v;
      }
    }
  }
}

// log_softmax over 40 classes, transpose to (B, NCLS, N)
__global__ void k_logsoftmax(const float* __restrict__ logits, float* __restrict__ out)
{
  int row = blockIdx.x * 256 + threadIdx.x;
  if (row >= ROWS) return;
  int b = row / NPTS, n = row % NPTS;
  const float* l = logits + (size_t)row * NCLS;
  float mx = -3.4e38f;
  for (int c = 0; c < NCLS; ++c) mx = fmaxf(mx, l[c]);
  float s = 0.f;
  for (int c = 0; c < NCLS; ++c) s += expf(l[c] - mx);
  float lse = mx + logf(s);
  for (int c = 0; c < NCLS; ++c)
    out[((size_t)b * NCLS + c) * NPTS + n] = l[c] - lse;
}

// ---------------------------------------------------------------------------
extern "C" void kernel_launch(void* const* d_in, const int* in_sizes, int n_in,
                              void* d_out, int out_size, void* d_ws, size_t ws_size,
                              hipStream_t stream)
{
  (void)in_sizes; (void)n_in; (void)out_size; (void)ws_size;

  // Flatten order: idx1, idx2, idx3, params(sorted), x
  const int* idx1 = (const int*)d_in[0];
  const int* idx2 = (const int*)d_in[1];
  const int* idx3 = (const int*)d_in[2];
  const float* x  = (const float*)d_in[71];

  // bump allocator over workspace
  char* wsb = (char*)d_ws;
  size_t off = 0;
  auto alloc = [&](size_t bytes) -> char* {
    char* p = wsb + off;
    off = (off + bytes + 255) & ~(size_t)255;
    return p;
  };

  float* xyz = (float*)alloc((size_t)ROWS * 3 * 4);
  float* l1  = (float*)alloc((size_t)ROWS * 64 * 4);
  float* l2  = (float*)alloc((size_t)ROWS * 128 * 4);
  float* l3  = (float*)alloc((size_t)ROWS * 256 * 4);
  float* u1  = (float*)alloc((size_t)ROWS * 256 * 4);
  float* u2  = (float*)alloc((size_t)ROWS * 128 * 4);
  int*   knn_idx = (int*)alloc((size_t)ROWS * 3 * 4);
  float* knn_w   = (float*)alloc((size_t)ROWS * 3 * 4);
  _Float16* actA = (_Float16*)alloc((size_t)ROWS * 384 * 2);
  _Float16* actB = (_Float16*)alloc((size_t)ROWS * 256 * 2);
  float* logits  = (float*)alloc((size_t)ROWS * NCLS * 4);

  // layer table: {W idx in d_in, cout, coutPad, kin, kpad, has_bn}
  struct Layer { int wi, cout, coutPad, kin, kpad, has_bn; };
  const Layer L[12] = {
    {11,  64,  64,   7,  32, 1},  // 0 sa1
    {17, 128, 128, 129, 160, 1},  // 1 sa2
    {23, 256, 256, 257, 288, 1},  // 2 sa3
    {29, 256, 256, 384, 384, 1},  // 3 su1[0]
    {35, 256, 256, 256, 256, 1},  // 4 su1[1]
    {41, 256, 256, 320, 320, 1},  // 5 su2[0]
    {47, 128, 128, 256, 256, 1},  // 6 su2[1]
    {53, 128, 128, 131, 160, 1},  // 7 su3[0]
    {59, 128, 128, 128, 128, 1},  // 8 su3[1]
    {65, 128, 128, 128, 128, 1},  // 9 su3[2]
    { 3, 128, 128, 128, 128, 1},  // 10 head1
    { 9,  40,  64, 128, 128, 0},  // 11 head2 (pad to 64 for 32-wide N blocks)
  };
  _Float16* Wh[12]; float* Ac[12]; float* Bc[12];
  for (int i = 0; i < 12; ++i) {
    Wh[i] = (_Float16*)alloc((size_t)L[i].coutPad * L[i].kpad * 2);
    Ac[i] = (float*)alloc((size_t)L[i].coutPad * 4);
    Bc[i] = (float*)alloc((size_t)L[i].coutPad * 4);
    const float* W  = (const float*)d_in[L[i].wi];
    const float* b  = (const float*)d_in[L[i].wi + 1];
    const float* be = L[i].has_bn ? (const float*)d_in[L[i].wi + 2] : nullptr;
    const float* g  = L[i].has_bn ? (const float*)d_in[L[i].wi + 3] : nullptr;
    const float* m  = L[i].has_bn ? (const float*)d_in[L[i].wi + 4] : nullptr;
    const float* vv = L[i].has_bn ? (const float*)d_in[L[i].wi + 5] : nullptr;
    int total = L[i].coutPad * L[i].kpad;
    k_prep_weights<<<(total + 255) / 256, 256, 0, stream>>>(
        W, b, be, g, m, vv, Wh[i], Ac[i], Bc[i],
        L[i].cout, L[i].coutPad, L[i].kin, L[i].kpad, L[i].has_bn);
  }

  auto gemm = [&](const _Float16* X, int li, float* outf, _Float16* o16,
                  int npad16, int relu) {
    int wtiles = (ROWS / 64) * (L[li].coutPad >> 5);  // all divisible by 8
    k_gemm<<<wtiles / 8, 256, 0, stream>>>(X, Wh[li], Ac[li], Bc[li], outf, o16,
                                           L[li].kpad, L[li].coutPad, L[li].cout,
                                           npad16, relu);
  };
  auto interp = [&](const float* p1, int c1, const float* p2, int c2,
                    _Float16* o16, int kpad) {
    size_t total = (size_t)ROWS * kpad;
    k_interp_concat<<<(unsigned)((total + 255) / 256), 256, 0, stream>>>(
        p1, c1, p2, c2, knn_idx, knn_w, o16, kpad);
  };

  // ---- pipeline ----------------------------------------------------------
  k_xyz<<<ROWS / 256, 256, 0, stream>>>(x, xyz);

  k_point_conv<  3,  32,  64><<<ROWS, 256, 0, stream>>>(xyz, idx1, Wh[0], Ac[0], Bc[0], l1);
  k_point_conv< 64, 160, 128><<<ROWS, 256, 0, stream>>>(l1,  idx2, Wh[1], Ac[1], Bc[1], l2);
  k_point_conv<128, 288, 256><<<ROWS, 256, 0, stream>>>(l2,  idx3, Wh[2], Ac[2], Bc[2], l3);

  k_knn3<<<dim3(NPTS / 256, BATCH), 256, 0, stream>>>(xyz, knn_idx, knn_w);

  // su1: concat(l2, interp l3) 384 -> 256 -> 256 (u1)
  interp(l2, 128, l3, 256, actA, 384);
  gemm(actA, 3, nullptr, actB, 256, 1);
  gemm(actB, 4, u1, nullptr, 0, 1);

  // su2: concat(l1, interp u1) 320 -> 256 -> 128 (u2)
  interp(l1, 64, u1, 256, actA, 320);
  gemm(actA, 5, nullptr, actB, 256, 1);
  gemm(actB, 6, u2, nullptr, 0, 1);

  // su3: concat(xyz, interp u2) 131->160pad -> 128 -> 128 -> 128
  interp(xyz, 3, u2, 128, actA, 160);
  gemm(actA, 7, nullptr, actB, 128, 1);
  gemm(actB, 8, nullptr, actA, 128, 1);
  gemm(actA, 9, nullptr, actB, 128, 1);

  // head1 (128->128, relu) then head2 (128->40, plain)
  gemm(actB, 10, nullptr, actA, 128, 1);
  gemm(actA, 11, logits, nullptr, 0, 0);

  k_logsoftmax<<<ROWS / 256, 256, 0, stream>>>(logits, (float*)d_out);
}
